// ChebyKANLayer_43241730736929
// MI455X (gfx1250) — matrix operand order repaired and checked by reading, not compiled, not run on hardware
//
#include <hip/hip_runtime.h>
#include <hip/hip_bf16.h>
#include <math.h>

typedef __attribute__((ext_vector_type(16))) _Float16 v16h;
typedef __attribute__((ext_vector_type(8)))  _Float16 v8h;
typedef __attribute__((ext_vector_type(8)))  float    v8f;

#define B_ROWS  16384
#define IN_DIM  1024
#define OUT_DIM 1024
#define NDEG    9        // D+1
#define BM      128      // output tile rows per block
#define BN      128      // output tile cols per block
#define A_STRIDE 40      // padded LDS row stride (halves): conflict-free 16B frag reads

// ---------------------------------------------------------------------------
// Pre-pass: repack coeffs (I, O, 9) fp32 -> W3[d][ib][o][ki] fp16, where
// i = ib*32 + ki.  This gives the per-lane-contiguous k order the WMMA
// B-fragments here use (A uses the same): lane reads 16 consecutive K (ki) values for its column o.
// One thread per (i, o): reads 9 contiguous floats, writes 9 scattered halves.
// ---------------------------------------------------------------------------
__global__ void repack_coeffs(const float* __restrict__ coeffs,
                              _Float16* __restrict__ w3) {
  int tid = blockIdx.x * blockDim.x + threadIdx.x;   // 0 .. I*O-1
  if (tid >= IN_DIM * OUT_DIM) return;
  int o = tid & (OUT_DIM - 1);
  int i = tid >> 10;
  int ib = i >> 5;
  int ki = i & 31;
  const float* src = coeffs + (size_t)i * (OUT_DIM * NDEG) + (size_t)o * NDEG;
#pragma unroll
  for (int d = 0; d < NDEG; ++d) {
    w3[(((size_t)(d * 32 + ib) * OUT_DIM) + o) * 32 + ki] = (_Float16)src[d];
  }
}

// ---------------------------------------------------------------------------
// Main fused kernel: y = Basis(16384 x 9216) @ W(9216 x 1024) via WMMA.
// Basis tiles (128 x 32 fp16) are generated on the fly in LDS from the
// Chebyshev recurrence T_{d+1} = 2 t T_d - T_{d-1}, t = tanh(x).
// 8 waves: wave = (wm 0..3) x (wn 0..1); each wave owns 32x64 of C
// as a 2x4 grid of 16x16 f32 accumulators.
// ---------------------------------------------------------------------------
__launch_bounds__(256, 1)
__global__ void cheby_kan_gemm(const float* __restrict__ x,
                               const _Float16* __restrict__ w3,
                               float* __restrict__ out) {
  __shared__ _Float16 Alds[BM * A_STRIDE];

  const int tid  = threadIdx.x;
  const int lane = tid & 31;
  const int wave = tid >> 5;
  const int wm   = wave & 3;          // M sub-block of 32 rows
  const int wn   = wave >> 2;         // N sub-block of 64 cols
  const int bm0  = blockIdx.y * BM;
  const int o0   = blockIdx.x * BN;

  // cooperative A-tile production: each thread owns 16 elements
  const int arow  = tid >> 1;          // 0..127
  const int acol0 = (tid & 1) * 16;    // 0 or 16

  // WMMA per-lane fragment addressing (contiguous k = 16*lhi + s, same for A and B)
  const int llo    = lane & 15;
  const int lhi    = lane >> 4;        // 0 or 1
  const int a_koff = lhi * 16;         // A: K {0..15} / {16..31}
  const int b_koff = lhi * 16;         // B: K {0..15} / {16..31}

  v8f acc[2][4];
#pragma unroll
  for (int m = 0; m < 2; ++m)
#pragma unroll
    for (int n = 0; n < 4; ++n) acc[m][n] = {};

  for (int ib = 0; ib < IN_DIM / 32; ++ib) {
    // ---- load x chunk, compute t = tanh(x); keep recurrence in registers
    float t[16], Tp[16], Tc[16];
    const float* xr = x + (size_t)(bm0 + arow) * IN_DIM + ib * 32 + acol0;
#pragma unroll
    for (int j = 0; j < 16; ++j) t[j] = tanhf(xr[j]);
#pragma unroll
    for (int j = 0; j < 16; ++j) { Tp[j] = 1.0f; Tc[j] = t[j]; }

#pragma unroll
    for (int d = 0; d < NDEG; ++d) {
      __syncthreads();   // previous K-block's A reads are done
      // store T_d tile (d==0 -> ones) as fp16
#pragma unroll
      for (int j = 0; j < 16; ++j) {
        float v = (d == 0) ? 1.0f : Tc[j];
        Alds[arow * A_STRIDE + acol0 + j] = (_Float16)v;
      }
      if (d >= 1) {      // advance recurrence for next degree
#pragma unroll
        for (int j = 0; j < 16; ++j) {
          float nx = __builtin_fmaf(2.0f * t[j], Tc[j], -Tp[j]);
          Tp[j] = Tc[j];
          Tc[j] = nx;
        }
      }
      __syncthreads();   // A tile visible to all waves

      // ---- B fragments straight from L2-resident repacked weights
      const _Float16* wblk =
          w3 + ((size_t)(d * 32 + ib) * OUT_DIM + (o0 + wn * 64)) * 32;
      // prefetch next degree's block (global_prefetch_b8)
      __builtin_prefetch(wblk + (size_t)32 * OUT_DIM * 32, 0, 1);

      v16h bfrag[4];
#pragma unroll
      for (int n = 0; n < 4; ++n) {
        const _Float16* p = wblk + (size_t)(n * 16 + llo) * 32 + b_koff;
        bfrag[n] = *(const v16h*)p;    // 32B, 32B-aligned
      }

      // ---- 2 (M) x 4 (N) WMMAs for this K=32 block
#pragma unroll
      for (int m = 0; m < 2; ++m) {
        const _Float16* ap = &Alds[(wm * 32 + m * 16 + llo) * A_STRIDE + a_koff];
        v8h a0 = *(const v8h*)(ap);
        v8h a1 = *(const v8h*)(ap + 8);
        v16h afrag = __builtin_shufflevector(a0, a1,
            0, 1, 2, 3, 4, 5, 6, 7, 8, 9, 10, 11, 12, 13, 14, 15);
#pragma unroll
        for (int n = 0; n < 4; ++n) {
          acc[m][n] = __builtin_amdgcn_wmma_f32_16x16x32_f16(
              false, afrag, false, bfrag[n], (short)0, acc[m][n], false, false);
        }
      }
    }
  }

  // ---- epilogue: C/D layout -> VGPR v holds rows v (lanes 0-15) and v+8
#pragma unroll
  for (int m = 0; m < 2; ++m) {
#pragma unroll
    for (int n = 0; n < 4; ++n) {
#pragma unroll
      for (int v = 0; v < 8; ++v) {
        int row = bm0 + wm * 32 + m * 16 + lhi * 8 + v;
        int col = o0 + wn * 64 + n * 16 + llo;
        out[(size_t)row * OUT_DIM + col] = acc[m][n][v];
      }
    }
  }
}

extern "C" void kernel_launch(void* const* d_in, const int* in_sizes, int n_in,
                              void* d_out, int out_size, void* d_ws, size_t ws_size,
                              hipStream_t stream) {
  const float* x      = (const float*)d_in[0];
  const float* coeffs = (const float*)d_in[1];
  float* out          = (float*)d_out;
  _Float16* w3        = (_Float16*)d_ws;   // needs 9*1024*1024*2 = ~18.9 MB

  repack_coeffs<<<(IN_DIM * OUT_DIM + 255) / 256, 256, 0, stream>>>(coeffs, w3);

  dim3 grid(OUT_DIM / BN, B_ROWS / BM);    // 8 x 128 blocks
  cheby_kan_gemm<<<grid, 256, 0, stream>>>(x, w3, out);
}
